// GATResidualBlock_24369644437898
// MI455X (gfx1250) — matrix operand order repaired and checked by reading, not compiled, not run on hardware
//
#include <hip/hip_runtime.h>
#include <hip/hip_bf16.h>

typedef __attribute__((ext_vector_type(16))) _Float16 v16h;
typedef __attribute__((ext_vector_type(8)))  float    v8f;
typedef __attribute__((ext_vector_type(8)))  unsigned v8u;

static constexpr int N_NODES = 100000;
static constexpr int N_EDGES = 1600000;
static constexpr int D       = 128;      // D_IN == D_OUT
static constexpr int HEADS   = 4;
static constexpr int CPH     = 32;       // channels per head
static constexpr float NEG_SLOPE = 0.2f;
static constexpr float LN_EPS    = 1e-5f;

union FragU { v8u u; v16h h; };

// monotonic float<->uint order mapping for atomic float-max
__device__ __forceinline__ unsigned f2ord(float f) {
    unsigned b = __float_as_uint(f);
    return (b >> 31) ? ~b : (b | 0x80000000u);
}
__device__ __forceinline__ float ord2f(unsigned o) {
    unsigned b = (o >> 31) ? (o ^ 0x80000000u) : ~o;
    return __uint_as_float(b);
}

// ---------------------------------------------------------------------------
// Kernel 1: h = x @ W via v_wmma_f32_16x16x32_f16.
// Block = 256 threads (8 waves). Each block computes 128 rows of h.
// W (f16, transposed) and the x row-tile (f16) staged in LDS, padded stride.
// ---------------------------------------------------------------------------
static constexpr int LDS_STRIDE = 132;   // halves; 66 dwords -> bank-conflict-free strided reads

__global__ __launch_bounds__(256) void gat_gemm_wmma(
    const float* __restrict__ x, const float* __restrict__ W,
    float* __restrict__ h, int n_nodes)
{
    __shared__ _Float16 Xh[128 * LDS_STRIDE];   // [row][k]
    __shared__ _Float16 WhT[128 * LDS_STRIDE];  // [n][k]  (transposed W)

    const int tid = threadIdx.x;
    const int row0 = blockIdx.x * 128;

    // Stage W transposed (coalesced global read, f32 -> f16)
    #pragma unroll
    for (int i = tid; i < 128 * 128; i += 256) {
        int k = i >> 7, n = i & 127;
        WhT[n * LDS_STRIDE + k] = (_Float16)W[i];        // W is [k][n] row-major
    }
    // Stage 128 rows of x (zero-pad past N)
    #pragma unroll
    for (int i = tid; i < 128 * 128; i += 256) {
        int r = i >> 7, k = i & 127;
        int row = row0 + r;
        Xh[r * LDS_STRIDE + k] = (row < n_nodes) ? (_Float16)x[(size_t)row * D + k]
                                                 : (_Float16)0.0f;
    }
    __syncthreads();

    const int wave = tid >> 5;
    const int lane = tid & 31;
    const int wrow = wave * 16;                 // 16-row tile per wave
    if (row0 + wrow >= n_nodes) return;         // N % 16 == 0 -> full tiles only

    const int mrow = lane & 15;
    const int half = lane >> 4;

    v8f acc[8] = {};                            // 8 N-tiles of 16 cols

    #pragma unroll
    for (int kt = 0; kt < 4; ++kt) {            // K = 128 in steps of 32
        // A fragment: row (wrow+mrow), documented split-K layout
        const unsigned* xu =
            (const unsigned*)(Xh + (wrow + mrow) * LDS_STRIDE + kt * 32);
        FragU a;
        #pragma unroll
        for (int v = 0; v < 8; ++v)
            a.u[v] = xu[(v >> 2) * 8 + half * 4 + (v & 3)];

        #pragma unroll
        for (int nt = 0; nt < 8; ++nt) {
            // B fragment: col nt*16 + (lane&15), split-K layout matching A (WhT)
            const unsigned* wu =
                (const unsigned*)(WhT + (nt * 16 + (lane & 15)) * LDS_STRIDE + kt * 32);
            FragU b;
            #pragma unroll
            for (int v = 0; v < 8; ++v)
                b.u[v] = wu[(v >> 2) * 8 + half * 4 + (v & 3)];

            acc[nt] = __builtin_amdgcn_wmma_f32_16x16x32_f16(
                false, a.h, false, b.h, (short)0, acc[nt], false, false);
        }
    }

    // C/D layout: lane -> col = lane&15 ; vgpr r -> row = r + 8*(lane>>4)
    #pragma unroll
    for (int nt = 0; nt < 8; ++nt) {
        #pragma unroll
        for (int r = 0; r < 8; ++r) {
            int row = row0 + wrow + half * 8 + r;
            int col = nt * 16 + (lane & 15);
            h[(size_t)row * D + col] = acc[nt][r];
        }
    }
}

// ---------------------------------------------------------------------------
// Kernel 2: per-(node,head) attention dots a_src/a_dst
// ---------------------------------------------------------------------------
__global__ __launch_bounds__(256) void gat_att_dots(
    const float* __restrict__ h, const float* __restrict__ att_src,
    const float* __restrict__ att_dst,
    float* __restrict__ a_s, float* __restrict__ a_d)
{
    int t = blockIdx.x * 256 + threadIdx.x;     // t = node*HEADS + head
    if (t >= N_NODES * HEADS) return;
    int head = t & (HEADS - 1);
    int node = t >> 2;
    const float4* hp = (const float4*)(h + (size_t)node * D + head * CPH);
    const float4* as = (const float4*)(att_src + head * CPH);
    const float4* ad = (const float4*)(att_dst + head * CPH);
    float ss = 0.f, sd = 0.f;
    #pragma unroll
    for (int i = 0; i < CPH / 4; ++i) {
        float4 hv = hp[i], av = as[i], dv = ad[i];
        ss += hv.x * av.x + hv.y * av.y + hv.z * av.z + hv.w * av.w;
        sd += hv.x * dv.x + hv.y * dv.y + hv.z * dv.z + hv.w * dv.w;
    }
    a_s[t] = ss;
    a_d[t] = sd;
}

// ---------------------------------------------------------------------------
// Kernel 3: segment-max of leaky-relu edge logits (ordered-uint atomic max)
// ---------------------------------------------------------------------------
__global__ __launch_bounds__(256) void gat_edge_max(
    const long long* __restrict__ ei,
    const float* __restrict__ a_s, const float* __restrict__ a_d,
    unsigned* __restrict__ m_bits)
{
    int e = blockIdx.x * 256 + threadIdx.x;
    if (e >= N_EDGES) return;
    int s = (int)ei[e];
    int d = (int)ei[N_EDGES + e];
    #pragma unroll
    for (int hh = 0; hh < HEADS; ++hh) {
        float ev = a_s[s * HEADS + hh] + a_d[d * HEADS + hh];
        ev = (ev >= 0.f) ? ev : NEG_SLOPE * ev;
        atomicMax(&m_bits[d * HEADS + hh], f2ord(ev));
    }
}

// Kernel 4: decode max (0 sentinel == no incoming edges -> 0.0)
__global__ __launch_bounds__(256) void gat_decode_max(
    const unsigned* __restrict__ m_bits, float* __restrict__ m)
{
    int t = blockIdx.x * 256 + threadIdx.x;
    if (t >= N_NODES * HEADS) return;
    unsigned o = m_bits[t];
    m[t] = (o == 0u) ? 0.f : ord2f(o);
}

// ---------------------------------------------------------------------------
// Kernel 5: denominator s = segment_sum(exp(e - m[dst]))
// ---------------------------------------------------------------------------
__global__ __launch_bounds__(256) void gat_edge_expsum(
    const long long* __restrict__ ei,
    const float* __restrict__ a_s, const float* __restrict__ a_d,
    const float* __restrict__ m, float* __restrict__ ssum)
{
    int e = blockIdx.x * 256 + threadIdx.x;
    if (e >= N_EDGES) return;
    int s = (int)ei[e];
    int d = (int)ei[N_EDGES + e];
    #pragma unroll
    for (int hh = 0; hh < HEADS; ++hh) {
        float ev = a_s[s * HEADS + hh] + a_d[d * HEADS + hh];
        ev = (ev >= 0.f) ? ev : NEG_SLOPE * ev;
        atomicAdd(&ssum[d * HEADS + hh], __expf(ev - m[d * HEADS + hh]));
    }
}

// ---------------------------------------------------------------------------
// Kernel 6: message pass. One wave per edge; lane l moves features [4l,4l+4).
// alpha recomputed (cheap) instead of storing a 25MB ex array.
// ---------------------------------------------------------------------------
__global__ __launch_bounds__(256) void gat_message(
    const long long* __restrict__ ei,
    const float* __restrict__ a_s, const float* __restrict__ a_d,
    const float* __restrict__ m, const float* __restrict__ ssum,
    const float* __restrict__ h, float* __restrict__ acc)
{
    int tid  = blockIdx.x * 256 + threadIdx.x;
    int e    = tid >> 5;
    int lane = tid & 31;
    if (e >= N_EDGES) return;
    int s = (int)ei[e];
    int d = (int)ei[N_EDGES + e];
    int head = lane >> 3;                       // 8 lanes (32 feats) per head

    float ev = a_s[s * HEADS + head] + a_d[d * HEADS + head];
    ev = (ev >= 0.f) ? ev : NEG_SLOPE * ev;
    float alpha = __expf(ev - m[d * HEADS + head]) /
                  (ssum[d * HEADS + head] + 1e-16f);

    float4 hv = *(const float4*)(h + (size_t)s * D + lane * 4);
    float* op = acc + (size_t)d * D + lane * 4;
    atomicAdd(op + 0, hv.x * alpha);
    atomicAdd(op + 1, hv.y * alpha);
    atomicAdd(op + 2, hv.z * alpha);
    atomicAdd(op + 3, hv.w * alpha);
}

// ---------------------------------------------------------------------------
// Kernel 7: bias + LayerNorm + PReLU + residual. One wave per node (wave32).
// ---------------------------------------------------------------------------
__global__ __launch_bounds__(256) void gat_finalize(
    const float* __restrict__ acc, const float* __restrict__ x,
    const float* __restrict__ bias, const float* __restrict__ gamma,
    const float* __restrict__ beta, const float* __restrict__ prelu_w,
    float* __restrict__ out)
{
    int tid  = blockIdx.x * 256 + threadIdx.x;
    int node = tid >> 5;
    int lane = tid & 31;
    if (node >= N_NODES) return;

    float4 v = *(const float4*)(acc + (size_t)node * D + lane * 4);
    float4 b = *(const float4*)(bias + lane * 4);
    v.x += b.x; v.y += b.y; v.z += b.z; v.w += b.w;

    float lsum = v.x + v.y + v.z + v.w;
    float lsq  = v.x * v.x + v.y * v.y + v.z * v.z + v.w * v.w;
    #pragma unroll
    for (int off = 16; off >= 1; off >>= 1) {   // wave32 butterfly
        lsum += __shfl_xor(lsum, off, 32);
        lsq  += __shfl_xor(lsq,  off, 32);
    }
    float mean = lsum * (1.0f / D);
    float var  = lsq * (1.0f / D) - mean * mean;
    float inv  = rsqrtf(var + LN_EPS);
    float pw   = prelu_w[0];

    float4 g  = *(const float4*)(gamma + lane * 4);
    float4 be = *(const float4*)(beta  + lane * 4);
    float4 xr = *(const float4*)(x + (size_t)node * D + lane * 4);

    float4 o;
    o.x = (v.x - mean) * inv * g.x + be.x;
    o.y = (v.y - mean) * inv * g.y + be.y;
    o.z = (v.z - mean) * inv * g.z + be.z;
    o.w = (v.w - mean) * inv * g.w + be.w;
    o.x = (o.x >= 0.f ? o.x : pw * o.x) + xr.x;
    o.y = (o.y >= 0.f ? o.y : pw * o.y) + xr.y;
    o.z = (o.z >= 0.f ? o.z : pw * o.z) + xr.z;
    o.w = (o.w >= 0.f ? o.w : pw * o.w) + xr.w;

    *(float4*)(out + (size_t)node * D + lane * 4) = o;
}

// ---------------------------------------------------------------------------
extern "C" void kernel_launch(void* const* d_in, const int* in_sizes, int n_in,
                              void* d_out, int out_size, void* d_ws, size_t ws_size,
                              hipStream_t stream)
{
    const float*     x       = (const float*)d_in[0];
    const long long* ei      = (const long long*)d_in[1];   // int64 [2,E]
    const float*     W       = (const float*)d_in[2];
    const float*     att_src = (const float*)d_in[3];
    const float*     att_dst = (const float*)d_in[4];
    const float*     bias    = (const float*)d_in[5];
    const float*     gamma   = (const float*)d_in[6];
    const float*     beta    = (const float*)d_in[7];
    const float*     prelu   = (const float*)d_in[8];
    float*           out     = (float*)d_out;

    // workspace layout (all 16B aligned): h | a_s | a_d | m | [mb | ssum | acc] (zeroed)
    char* p = (char*)d_ws;
    float*    h    = (float*)p;    p += (size_t)N_NODES * D * 4;
    float*    a_s  = (float*)p;    p += (size_t)N_NODES * HEADS * 4;
    float*    a_d  = (float*)p;    p += (size_t)N_NODES * HEADS * 4;
    float*    m    = (float*)p;    p += (size_t)N_NODES * HEADS * 4;
    unsigned* mb   = (unsigned*)p; p += (size_t)N_NODES * HEADS * 4;
    float*    ssum = (float*)p;    p += (size_t)N_NODES * HEADS * 4;
    float*    acc  = (float*)p;    p += (size_t)N_NODES * D * 4;

    size_t zbytes = (size_t)N_NODES * HEADS * 4 * 2 + (size_t)N_NODES * D * 4;
    hipMemsetAsync(mb, 0, zbytes, stream);      // capturable stream op

    gat_gemm_wmma  <<<(N_NODES + 127) / 128, 256, 0, stream>>>(x, W, h, N_NODES);
    gat_att_dots   <<<(N_NODES * HEADS + 255) / 256, 256, 0, stream>>>(h, att_src, att_dst, a_s, a_d);
    gat_edge_max   <<<(N_EDGES + 255) / 256, 256, 0, stream>>>(ei, a_s, a_d, mb);
    gat_decode_max <<<(N_NODES * HEADS + 255) / 256, 256, 0, stream>>>(mb, m);
    gat_edge_expsum<<<(N_EDGES + 255) / 256, 256, 0, stream>>>(ei, a_s, a_d, m, ssum);
    gat_message    <<<(N_EDGES * 32 + 255) / 256, 256, 0, stream>>>(ei, a_s, a_d, m, ssum, h, acc);
    gat_finalize   <<<(N_NODES * 32 + 255) / 256, 256, 0, stream>>>(acc, x, bias, gamma, beta, prelu, out);
}
